// S2Attn_42932493090830
// MI455X (gfx1250) — matrix operand order repaired and checked by reading, not compiled, not run on hardware
//
#include <hip/hip_runtime.h>
#include <hip/hip_bf16.h>

typedef __attribute__((ext_vector_type(16))) _Float16 v16h;
typedef __attribute__((ext_vector_type(8)))  _Float16 v8h;
typedef __attribute__((ext_vector_type(8)))  float    v8f;

#define BATCH 2
#define SEQ   8192
#define CDIM  2048
#define HEADS 16
#define HD    128
#define GS    256
#define NGRP  32
#define QKV_ELEMS ((size_t)BATCH * HEADS * SEQ * HD)   // 33,554,432 halves = 64MB each

__device__ __forceinline__ void lds_fence() {
    asm volatile("s_wait_dscnt 0" ::: "memory");
}
__device__ __forceinline__ void wait_async() {
    asm volatile("s_wait_asynccnt 0" ::: "memory");
}
// Async DMA 16B global -> LDS (per-lane LDS destination), tracked by ASYNCcnt.
__device__ __forceinline__ void async_b128(unsigned lds_off, const void* gptr) {
    asm volatile("global_load_async_to_lds_b128 %0, %1, off"
                 :: "v"(lds_off), "v"(gptr)
                 : "memory");
}

// Build a 16-half A/B fragment from two 16B-aligned 8-half chunks.
__device__ __forceinline__ v16h ldfrag(const _Float16* p0, const _Float16* p1) {
    v8h a = *(const v8h*)p0;
    v8h b = *(const v8h*)p1;
    v16h r;
#pragma unroll
    for (int i = 0; i < 8; ++i) { r[i] = a[i]; r[8 + i] = b[i]; }
    return r;
}

// ---------------------------------------------------------------------------
// Kernel 1: fused QKV projection. out = x @ W^T + b, f16 into workspace.
//   Q (mat 0): [B][H][N][HD], roll(-128) pre-applied for h>=8, scaled 1/sqrt(HD)
//   K (mat 1): [B][H][N][HD], roll(-128) pre-applied for h>=8
//   V (mat 2): [B][H][G][HD][GS] (group-transposed so attention can async-copy it)
// Tile: 256 rows x 128 cols, K-chunks of 32, 256 threads; wave = 64x64 (4x4 WMMA).
// ---------------------------------------------------------------------------
__global__ __launch_bounds__(256) void s2_proj_kernel(
    const float* __restrict__ x,
    const float* __restrict__ Wq, const float* __restrict__ bq,
    const float* __restrict__ Wk, const float* __restrict__ bk,
    const float* __restrict__ Wv, const float* __restrict__ bv,
    _Float16* __restrict__ qs, _Float16* __restrict__ ks, _Float16* __restrict__ vs)
{
    __shared__ _Float16 Ax[256 * 40];   // x tile, f16, padded stride 40 halves (20KB)
    __shared__ _Float16 Wl[128 * 40];   // W tile, f16                          (10KB)

    const int mat = blockIdx.z;
    const float* __restrict__ W    = (mat == 0) ? Wq : ((mat == 1) ? Wk : Wv);
    const float* __restrict__ bias = (mat == 0) ? bq : ((mat == 1) ? bk : bv);
    _Float16* __restrict__ dst     = (mat == 0) ? qs : ((mat == 1) ? ks : vs);
    const float scale = (mat == 0) ? 0.08838834764831845f : 1.0f; // 1/sqrt(128) into Q

    const int r0 = blockIdx.x * 256;   // global row tile (over B*N = 16384)
    const int c0 = blockIdx.y * 128;   // output-channel tile (over C = 2048)
    const int tid  = threadIdx.x;
    const int lane = tid & 31;
    const int wave = tid >> 5;
    const int wm = wave & 3;           // 4 wave-rows  of 64
    const int wn = wave >> 2;          // 2 wave-cols  of 64
    const int lr = lane & 15;
    const int hi = lane >> 4;

    v8f acc[4][4] = {};

    for (int kb = 0; kb < CDIM; kb += 32) {
        // ---- stage x (256x32) and W (128x32), f32 -> f16, into LDS ----
#pragma unroll
        for (int i = 0; i < 8; ++i) {
            int f4  = tid + i * 256;       // 2048 float4 chunks
            int row = f4 >> 3;
            int cq  = (f4 & 7) * 4;
            float4 xa = *(const float4*)(x + (size_t)(r0 + row) * CDIM + kb + cq);
            _Float16* pa = Ax + row * 40 + cq;
            pa[0] = (_Float16)xa.x; pa[1] = (_Float16)xa.y;
            pa[2] = (_Float16)xa.z; pa[3] = (_Float16)xa.w;
        }
#pragma unroll
        for (int i = 0; i < 4; ++i) {
            int f4  = tid + i * 256;       // 1024 float4 chunks
            int row = f4 >> 3;
            int cq  = (f4 & 7) * 4;
            float4 wa = *(const float4*)(W + (size_t)(c0 + row) * CDIM + kb + cq);
            _Float16* pw = Wl + row * 40 + cq;
            pw[0] = (_Float16)wa.x; pw[1] = (_Float16)wa.y;
            pw[2] = (_Float16)wa.z; pw[3] = (_Float16)wa.w;
        }
        __syncthreads();

        // ---- fragments + 16 WMMAs ----
        v16h aF[4], bF[4];
#pragma unroll
        for (int mt = 0; mt < 4; ++mt) {
            const _Float16* p = Ax + (wm * 64 + mt * 16 + lr) * 40;
            aF[mt] = ldfrag(p + hi * 16, p + hi * 16 + 8);
        }
#pragma unroll
        for (int nt = 0; nt < 4; ++nt) {
            const _Float16* p = Wl + (wn * 64 + nt * 16 + lr) * 40 + hi * 16;
            bF[nt] = ldfrag(p, p + 8);
        }
#pragma unroll
        for (int mt = 0; mt < 4; ++mt)
#pragma unroll
            for (int nt = 0; nt < 4; ++nt)
                acc[mt][nt] = __builtin_amdgcn_wmma_f32_16x16x32_f16(
                    false, aF[mt], false, bF[nt], (short)0, acc[mt][nt], false, false);
        __syncthreads();
    }

    // ---- epilogue: bias, scale, shift-store as f16 ----
#pragma unroll
    for (int nt = 0; nt < 4; ++nt) {
        int co = c0 + wn * 64 + nt * 16 + lr;
        int h  = co >> 7;
        int d  = co & 127;
        float bv_ = bias[co];
        int sh = (h >= 8) ? (SEQ - 128) : 0;
#pragma unroll
        for (int mt = 0; mt < 4; ++mt) {
#pragma unroll
            for (int v = 0; v < 8; ++v) {
                int R  = r0 + wm * 64 + mt * 16 + v + hi * 8;
                int bb = R >> 13;
                int n  = R & (SEQ - 1);
                int n2 = (n + sh) & (SEQ - 1);
                float val = (acc[mt][nt][v] + bv_) * scale;
                size_t idx;
                if (mat == 2) {  // V: [b][h][g][d][key]
                    int gg  = n2 >> 8;
                    int key = n2 & (GS - 1);
                    idx = (((size_t)(bb * HEADS + h) * NGRP + gg) << 15)
                        + (size_t)d * GS + key;
                } else {         // Q/K: [b][h][n'][d]
                    idx = ((size_t)(bb * HEADS + h) * SEQ + n2) * HD + d;
                }
                dst[idx] = (_Float16)val;
            }
        }
    }
}

// ---------------------------------------------------------------------------
// Kernel 2: group-local causal attention, one WG per (b,h,g).
// 512 threads = 16 waves; wave w owns query rows [16w, 16w+16).
// K/V staged by async-to-LDS DMA; flash-style online softmax over 32-key chunks.
// ---------------------------------------------------------------------------
__global__ __launch_bounds__(512) void s2_attn_kernel(
    const _Float16* __restrict__ qs, const _Float16* __restrict__ ks,
    const _Float16* __restrict__ vs, float* __restrict__ out)
{
    __shared__ _Float16 Kl[256 * 136];   // K tile [key][d], padded    (69,632 B)
    __shared__ _Float16 Vt[128 * 264];   // V tile [d][key], padded    (67,584 B)
    __shared__ _Float16 Pl[16 * 640];    // per-wave P scratch         (20,480 B)

    const int g = blockIdx.x, h = blockIdx.y, b = blockIdx.z;
    const int tid  = threadIdx.x;
    const int lane = tid & 31;
    const int wave = tid >> 5;
    const int lr = lane & 15;
    const int hi = lane >> 4;

    const size_t qkbase = ((size_t)(b * HEADS + h) * SEQ + g * GS) * HD;
    const _Float16* ksg = ks + qkbase;                                    // [key][128]
    const _Float16* vsg = vs + (((size_t)(b * HEADS + h) * NGRP + g) << 15); // [d][256]

    // ---- async DMA stage K (row-padded) ----
#pragma unroll
    for (int i = 0; i < 8; ++i) {
        int idx = tid + i * 512;            // 4096 chunks of 8 halves
        int key = idx >> 4;
        int cc  = (idx & 15) * 8;
        async_b128((unsigned)(size_t)(Kl + key * 136 + cc), ksg + key * HD + cc);
    }
    // ---- async DMA stage V (already transposed in workspace) ----
#pragma unroll
    for (int i = 0; i < 8; ++i) {
        int idx = tid + i * 512;
        int d   = idx >> 5;
        int cs  = (idx & 31) * 8;
        async_b128((unsigned)(size_t)(Vt + d * 264 + cs), vsg + d * GS + cs);
    }

    // ---- load Q A-fragments (already shifted + pre-scaled by 1/sqrt(hd)) ----
    const int qb = wave * 16;
    const _Float16* qp = qs + qkbase + (size_t)(qb + lr) * HD;
    v16h qF[4];
#pragma unroll
    for (int cf = 0; cf < 4; ++cf)
        qF[cf] = ldfrag(qp + cf * 32 + hi * 16, qp + cf * 32 + hi * 16 + 8);

    wait_async();
    __syncthreads();

    v8f acc[8] = {};
    float m8[8], l8[8];
#pragma unroll
    for (int v = 0; v < 8; ++v) { m8[v] = -1e30f; l8[v] = 0.0f; }

    _Float16* Pw = Pl + wave * 640;
    const int nch = (wave >> 1) + 1;        // causal: skip fully-masked chunks

    for (int ch = 0; ch < nch; ++ch) {
        const int kk = ch * 32;

        // ---- S = Q K^T  (two 16x16 tiles, K-dim = 128 in 4 steps) ----
        v8f S0 = {}, S1 = {};
#pragma unroll
        for (int cf = 0; cf < 4; ++cf) {
            const _Float16* kp0 = Kl + (kk + lr) * 136 + cf * 32 + hi * 16;
            v16h b0 = ldfrag(kp0, kp0 + 8);
            S0 = __builtin_amdgcn_wmma_f32_16x16x32_f16(false, qF[cf], false, b0,
                                                        (short)0, S0, false, false);
            const _Float16* kp1 = Kl + (kk + 16 + lr) * 136 + cf * 32 + hi * 16;
            v16h b1 = ldfrag(kp1, kp1 + 8);
            S1 = __builtin_amdgcn_wmma_f32_16x16x32_f16(false, qF[cf], false, b1,
                                                        (short)0, S1, false, false);
        }

        // ---- causal mask + online softmax (row stats via width-16 shuffles) ----
        float p0[8], p1[8];
#pragma unroll
        for (int v = 0; v < 8; ++v) {
            int row = qb + v + hi * 8;
            float s0 = ((kk + lr)      <= row) ? S0[v] : -1e9f;
            float s1 = ((kk + 16 + lr) <= row) ? S1[v] : -1e9f;
            float mx = fmaxf(s0, s1);
            mx = fmaxf(mx, __shfl_xor(mx, 1, 16));
            mx = fmaxf(mx, __shfl_xor(mx, 2, 16));
            mx = fmaxf(mx, __shfl_xor(mx, 4, 16));
            mx = fmaxf(mx, __shfl_xor(mx, 8, 16));
            float nm = fmaxf(m8[v], mx);
            float sc = __expf(m8[v] - nm);
            float e0 = __expf(s0 - nm);
            float e1 = __expf(s1 - nm);
            float rs = e0 + e1;
            rs += __shfl_xor(rs, 1, 16);
            rs += __shfl_xor(rs, 2, 16);
            rs += __shfl_xor(rs, 4, 16);
            rs += __shfl_xor(rs, 8, 16);
            l8[v] = l8[v] * sc + rs;
            m8[v] = nm;
            p0[v] = e0; p1[v] = e1;
#pragma unroll
            for (int t = 0; t < 8; ++t) acc[t][v] *= sc;
        }

        // ---- P: C-layout -> A-layout via wave-private LDS ----
#pragma unroll
        for (int v = 0; v < 8; ++v) {
            Pw[(v + hi * 8) * 40 + lr]      = (_Float16)p0[v];
            Pw[(v + hi * 8) * 40 + 16 + lr] = (_Float16)p1[v];
        }
        lds_fence();
        v16h pA = ldfrag(Pw + lr * 40 + hi * 16, Pw + lr * 40 + hi * 16 + 8);

        // ---- O += P V  (8 channel tiles) ----
#pragma unroll
        for (int t = 0; t < 8; ++t) {
            const _Float16* vp = Vt + (t * 16 + lr) * 264 + kk + hi * 16;
            v16h bV = ldfrag(vp, vp + 8);
            acc[t] = __builtin_amdgcn_wmma_f32_16x16x32_f16(false, pA, false, bV,
                                                            (short)0, acc[t], false, false);
        }
    }

    // ---- normalize + un-shift + store f32 output [B][N][C] ----
    const int sh = (h >= 8) ? 128 : 0;
#pragma unroll
    for (int v = 0; v < 8; ++v) {
        float inv = 1.0f / l8[v];
        int row   = g * GS + qb + v + hi * 8;
        int n_out = (row + sh) & (SEQ - 1);
        float* op = out + ((size_t)b * SEQ + n_out) * CDIM + h * HD + lr;
#pragma unroll
        for (int t = 0; t < 8; ++t)
            op[t * 16] = acc[t][v] * inv;
    }
}

extern "C" void kernel_launch(void* const* d_in, const int* in_sizes, int n_in,
                              void* d_out, int out_size, void* d_ws, size_t ws_size,
                              hipStream_t stream) {
    const float* x  = (const float*)d_in[0];
    const float* Wq = (const float*)d_in[1];
    const float* bq = (const float*)d_in[2];
    const float* Wk = (const float*)d_in[3];
    const float* bk = (const float*)d_in[4];
    const float* Wv = (const float*)d_in[5];
    const float* bv = (const float*)d_in[6];

    _Float16* qs = (_Float16*)d_ws;
    _Float16* ks = qs + QKV_ELEMS;
    _Float16* vs = ks + QKV_ELEMS;

    dim3 g1((BATCH * SEQ) / 256, CDIM / 128, 3);   // 64 x 16 x 3
    s2_proj_kernel<<<g1, 256, 0, stream>>>(x, Wq, bq, Wk, bk, Wv, bv, qs, ks, vs);

    dim3 g2(NGRP, HEADS, BATCH);                   // 32 x 16 x 2
    s2_attn_kernel<<<g2, 512, 0, stream>>>(qs, ks, vs, (float*)d_out);
}
